// adaConv2d_63711544869421
// MI455X (gfx1250) — matrix operand order repaired and checked by reading, not compiled, not run on hardware
//
#include <hip/hip_runtime.h>

typedef _Float16 f16;
typedef __attribute__((ext_vector_type(16))) _Float16 v16h;
typedef __attribute__((ext_vector_type(8)))  _Float16 v8h;
typedef __attribute__((ext_vector_type(2)))  _Float16 v2h;
typedef __attribute__((ext_vector_type(8)))  float    v8f;
typedef __attribute__((ext_vector_type(4)))  int      v4i;

#define CIN   256
#define COUT  256
#define HW    64
#define P_    (HW*HW)
#define NTHREADS 512

// LDS layout (bytes). Strides padded (+8 f16) so 16-lane b128 fragment loads
// walk banks 4m..4m+3 -> conflict-free.
#define A_STRIDE 264
#define B_STRIDE 264
#define OUT_STRIDE 68
#define LDS_A_OFF  0
#define LDS_B0_OFF (64*A_STRIDE*2)                 /* 33792  */
#define LDS_B1_OFF (LDS_B0_OFF + 256*B_STRIDE*2)   /* 168960 */
#define LDS_F_OFF  (LDS_B1_OFF + 256*B_STRIDE*2)   /* 304128 */
#define LDS_S_OFF  (LDS_F_OFF + 64*4)
#define LDS_TOTAL  (LDS_S_OFF + 64*4)              /* 304640 <= 320K WGP */

#if defined(__has_builtin)
#  if __has_builtin(__builtin_amdgcn_global_load_async_to_lds_b128)
#    define HAVE_ASYNC_LDS 1
#  endif
#  if __has_builtin(__builtin_amdgcn_s_wait_asynccnt)
#    define WAIT_ASYNC(n) __builtin_amdgcn_s_wait_asynccnt(n)
#  else
#    define WAIT_ASYNC(n) asm volatile("s_wait_asynccnt %0" :: "i"(n) : "memory")
#  endif
#endif

__device__ __forceinline__ void stage_b_tap(const f16* wt, f16* Bdst, int t, int tid) {
  const f16* wsrc = wt + (size_t)t * (COUT * CIN);
#pragma unroll
  for (int it = 0; it < 16; ++it) {
    int idx = tid + (it << 9);          // 0..8191 chunks of 8 f16
    int o  = idx >> 5;
    int cp = idx & 31;
#if defined(HAVE_ASYNC_LDS)
    __builtin_amdgcn_global_load_async_to_lds_b128(
        (v4i*)(void*)(wsrc + o * CIN + cp * 8),
        (v4i*)(void*)(Bdst + o * B_STRIDE + cp * 8), 0, 0);
#else
    *(v8h*)(Bdst + o * B_STRIDE + cp * 8) = *(const v8h*)(wsrc + o * CIN + cp * 8);
#endif
  }
}

// weight (Cout,Cin,3,3) f32  ->  Wt[t][o][c] f16   (t = ki*3+kj)
__global__ void adaconv_wtrans(const float* __restrict__ w, f16* __restrict__ wt) {
  int idx = blockIdx.x * blockDim.x + threadIdx.x;
  if (idx >= 9 * COUT * CIN) return;
  int c = idx & 255;
  int o = (idx >> 8) & 255;
  int t = idx >> 16;
  wt[idx] = (f16)w[(o * CIN + c) * 9 + t];
}

__global__ __launch_bounds__(NTHREADS)
void adaconv_main(const float* __restrict__ x, const float* __restrict__ scales,
                  const f16* __restrict__ wt, const float* __restrict__ bias,
                  float* __restrict__ out) {
  extern __shared__ __align__(16) char smem[];
  f16*   As = (f16*)(smem + LDS_A_OFF);
  f16*   B0 = (f16*)(smem + LDS_B0_OFF);
  f16*   B1 = (f16*)(smem + LDS_B1_OFF);
  float* Fl = (float*)(smem + LDS_F_OFF);
  int*   Sl = (int*)(smem + LDS_S_OFF);

  const int tid = threadIdx.x;
  const int bb  = blockIdx.x >> 6;   // batch
  const int h   = blockIdx.x & 63;   // image row; pixels = (h, w=0..63)

  // ---- per-pixel scale params: sm = 3x3 zero-padded mean; s = ceil; f = sm/s
  if (tid < 64) {
    const int w = tid;
    float ssum = 0.f;
    for (int dy = -1; dy <= 1; ++dy)
      for (int dx = -1; dx <= 1; ++dx) {
        int yy = h + dy, xx = w + dx;
        if (yy >= 0 && yy < HW && xx >= 0 && xx < HW)
          ssum += scales[bb * P_ + yy * HW + xx];
      }
    float sm = ssum * (1.0f / 9.0f);
    int s = (int)ceilf(sm);
    s = s < 1 ? 1 : (s > 3 ? 3 : s);
    Fl[w] = sm / (float)s;
    Sl[w] = s;
  }

  // prologue: start streaming tap-0 weights while params settle
  stage_b_tap(wt, B0, 0, tid);
  __syncthreads();

  const int lane = tid & 31;
  const int wid  = tid >> 5;   // 0..15
  const int mt   = wid & 3;    // 16-pixel M tile (0..3)
  const int ng   = wid >> 2;   // 64-output N group (0..3)
  const int hb   = lane >> 4;  // lane half
  const int ln   = lane & 15;

  const int   my_px = tid & 63;     // pixel (w coord) this thread builds
  const int   my_co = tid >> 6;     // channel pair sub-lane (0..7)
  const float f     = Fl[my_px];
  const int   s     = Sl[my_px];

  v8f acc[4];
#pragma unroll
  for (int i = 0; i < 4; ++i) acc[i] = (v8f){};

#pragma unroll 1
  for (int t = 0; t < 9; ++t) {
    const int ti = t / 3, tj = t % 3;
    f16* Bcur = (t & 1) ? B1 : B0;

    // ---- stream next tap's B tile into the other buffer (fully overlapped)
    if (t < 8) stage_b_tap(wt, (t & 1) ? B0 : B1, t + 1, tid);

    // ---- build resampled A tile for tap t: As[px][c] (f16), 2 channels/thread
    {
      int ya, yb; float wya, wyb;
      if (ti == 0)      { ya = h - s; yb = h;     wya = f;        wyb = 1.f - f; }
      else if (ti == 1) { ya = h;     yb = h;     wya = 1.f;      wyb = 0.f;     }
      else              { ya = h;     yb = h + s; wya = 1.f - f;  wyb = f;       }
      int xa, xc; float wxa, wxb;
      if (tj == 0)      { xa = my_px - s; xc = my_px;     wxa = f;       wxb = 1.f - f; }
      else if (tj == 1) { xa = my_px;     xc = my_px;     wxa = 1.f;     wxb = 0.f;     }
      else              { xa = my_px;     xc = my_px + s; wxa = 1.f - f; wxb = f;       }
      if (ya < 0 || ya >= HW) { wya = 0.f; ya = 0; }
      if (yb < 0 || yb >= HW) { wyb = 0.f; yb = 0; }
      if (xa < 0 || xa >= HW) { wxa = 0.f; xa = 0; }
      if (xc < 0 || xc >= HW) { wxb = 0.f; xc = 0; }
      const float w00 = wya * wxa, w01 = wya * wxb, w10 = wyb * wxa, w11 = wyb * wxb;
      const int oAA = ya * HW + xa, oAB = ya * HW + xc;
      const int oBA = yb * HW + xa, oBB = yb * HW + xc;
      // channels c0 = cb*16 + my_co*2  and  c0+1  (pair -> ds_store_b32)
      const float* xp = x + (size_t)(bb * CIN + my_co * 2) * P_;
      f16* arow = As + my_px * A_STRIDE + my_co * 2;
#pragma unroll 4
      for (int cb = 0; cb < 16; ++cb) {
        const float* x0 = xp;
        const float* x1 = xp + P_;
        float v0 = w00 * x0[oAA] + w01 * x0[oAB] + w10 * x0[oBA] + w11 * x0[oBB];
        float v1 = w00 * x1[oAA] + w01 * x1[oAB] + w10 * x1[oBA] + w11 * x1[oBB];
        v2h pr; pr.x = (f16)v0; pr.y = (f16)v1;
        *(v2h*)(arow + cb * 16) = pr;
        xp += 16 * P_;
      }
    }

#if defined(HAVE_ASYNC_LDS)
    // async loads complete in order: waiting <=16 drains tap t's chunks while
    // tap t+1's 16 chunks stay in flight.
    if (t < 8) WAIT_ASYNC(16); else WAIT_ASYNC(0);
#endif
    __syncthreads();

    // ---- GEMM for this tap: K = 256 channels = 8 wmma k-steps
#pragma unroll
    for (int kk = 0; kk < 8; ++kk) {
      v16h a;
      const f16* ap = As + (mt * 16 + ln) * A_STRIDE + kk * 32 + hb * 8;
      *((v8h*)&a)     = *(const v8h*)ap;          // elems 0..7 : K = hb*8+e
      *((v8h*)&a + 1) = *(const v8h*)(ap + 16);   // elems 8..15: K = 16+hb*8+e
      v16h bfr[4];
#pragma unroll
      for (int nt = 0; nt < 4; ++nt) {
        const f16* bp = Bcur + (ng * 64 + nt * 16 + ln) * B_STRIDE + kk * 32 + hb * 8;
        *((v8h*)&bfr[nt])     = *(const v8h*)bp;        // K = hb*8 + 0..7
        *((v8h*)&bfr[nt] + 1) = *(const v8h*)(bp + 16); // K = 16 + hb*8 + 0..7
      }
#pragma unroll
      for (int nt = 0; nt < 4; ++nt)
        acc[nt] = __builtin_amdgcn_wmma_f32_16x16x32_f16(
            false, a, false, bfr[nt], (short)0, acc[nt], false, false);
    }
    __syncthreads();   // protects A and the B buffer t+2 will overwrite
  }

  // ---- epilogue: transpose accumulators through LDS for coalesced stores
  float* Ol = (float*)(smem + LDS_B0_OFF);  // reuse B0 region (needs 69632 B)
#pragma unroll
  for (int nt = 0; nt < 4; ++nt)
#pragma unroll
    for (int r = 0; r < 8; ++r) {
      int o = ng * 64 + nt * 16 + ln;
      int m = mt * 16 + hb * 8 + r;
      Ol[o * OUT_STRIDE + m] = acc[nt][r];
    }
  __syncthreads();

  const size_t obase = ((size_t)bb * COUT) << 12;
  for (int ob = 0; ob < COUT; ob += 8) {
    int o = ob + my_co;
    out[obase + ((size_t)o << 12) + h * HW + my_px] =
        Ol[o * OUT_STRIDE + my_px] + bias[o];
  }
}

extern "C" void kernel_launch(void* const* d_in, const int* in_sizes, int n_in,
                              void* d_out, int out_size, void* d_ws, size_t ws_size,
                              hipStream_t stream) {
  const float* x      = (const float*)d_in[0];
  const float* scales = (const float*)d_in[1];
  const float* weight = (const float*)d_in[2];
  const float* bias   = (const float*)d_in[3];
  float* out = (float*)d_out;
  f16*   wtr = (f16*)d_ws;  // 9*256*256 f16 = 1,179,648 bytes

  (void)hipFuncSetAttribute((const void*)adaconv_main,
                            hipFuncAttributeMaxDynamicSharedMemorySize, LDS_TOTAL);

  const int wtot = 9 * COUT * CIN;
  adaconv_wtrans<<<(wtot + 255) / 256, 256, 0, stream>>>(weight, wtr);
  adaconv_main<<<256, NTHREADS, LDS_TOTAL, stream>>>(x, scales, wtr, bias, out);
}
